// ModulatedConv_44624710206002
// MI455X (gfx1250) — matrix operand order repaired and checked by reading, not compiled, not run on hardware
//
#include <hip/hip_runtime.h>

typedef __attribute__((ext_vector_type(16))) _Float16 v16h;
typedef __attribute__((ext_vector_type(8)))  _Float16 v8h;
typedef __attribute__((ext_vector_type(8)))  float    v8f;
typedef __attribute__((ext_vector_type(4)))  int      v4i;

#define B_      8
#define CIN_    64
#define COUT_   64
#define H_      256
#define W_      256
#define LATENT_ 512
#define EPS_    1e-8f
#define CPAD    72

// Workspace layout:
//   [0, 2048)        : style, B*CIN floats
//   [4096, 593920)   : packed A fragments (576 KB)
//   [1MB, 1MB+64MB)  : xT, f16 HWC copy of x  (only if ws_size allows)
#define XT_OFF   (1u << 20)
#define XT_BYTES ((size_t)B_ * H_ * W_ * CIN_ * 2)

#if __has_builtin(__builtin_amdgcn_global_load_async_to_lds_b128)
#define ASYNC_LDS 1
typedef __attribute__((address_space(1))) v4i* gas1_t;   // global int4*
typedef __attribute__((address_space(3))) v4i* las3_t;   // LDS int4*
#endif

// ---------------------------------------------------------------------------
// Kernel 1: style[b, cin] = w[b,:] . mod_w[cin,:] + mod_b[cin]
// ---------------------------------------------------------------------------
__global__ void style_kernel(const float* __restrict__ w,
                             const float* __restrict__ mod_w,
                             const float* __restrict__ mod_b,
                             float* __restrict__ style) {
    int b   = blockIdx.x;
    int cin = threadIdx.x;
    const float* wr = w + (size_t)b * LATENT_;
    const float* mr = mod_w + (size_t)cin * LATENT_;
    float s = 0.f;
    #pragma unroll 4
    for (int l = 0; l < LATENT_; ++l) s += wr[l] * mr[l];
    style[b * CIN_ + cin] = s + mod_b[cin];
}

// ---------------------------------------------------------------------------
// Kernel 2: modulate + demodulate, write f16 weights pre-packed in the
// WMMA 16x32 f16 A-fragment layout (ISA 7.12.2).
// ---------------------------------------------------------------------------
__global__ void pack_kernel(const float* __restrict__ weight,
                            const float* __restrict__ style,
                            _Float16* __restrict__ packedA) {
    int b    = blockIdx.x >> 6;   // grid = B*COUT
    int cout = blockIdx.x & 63;
    int cin  = threadIdx.x;       // 64 threads

    float st = style[b * CIN_ + cin];
    const float* wr = weight + ((size_t)cout * CIN_ + cin) * 9;

    float wv[9];
    float ss = 0.f;
    #pragma unroll
    for (int t = 0; t < 9; ++t) { wv[t] = wr[t] * st; ss += wv[t] * wv[t]; }

    __shared__ float red[64];
    red[cin] = ss;
    __syncthreads();
    #pragma unroll
    for (int off = 32; off > 0; off >>= 1) {
        if (cin < off) red[cin] += red[cin + off];
        __syncthreads();
    }
    float d = rsqrtf(red[0] + EPS_);

    int coutTile = cout >> 4, m = cout & 15;
    int chunk = cin >> 5, Kk = cin & 31;
    int hi = (Kk >> 3) & 1;
    int j  = (Kk & 7) + ((Kk & 16) ? 8 : 0);
    int lane = hi * 16 + m;
    #pragma unroll
    for (int t = 0; t < 9; ++t) {
        size_t addr = (((((size_t)b * 4 + coutTile) * 9 + t) * 2 + chunk) * 32 + lane) * 16 + j;
        packedA[addr] = (_Float16)(wv[t] * d);
    }
}

// ---------------------------------------------------------------------------
// Kernel 2b: transpose+convert x (NCHW f32) -> xT (NHWC f16) via LDS tile.
// Coalesced f32 reads, contiguous 4KB b128 writes per block.
// ---------------------------------------------------------------------------
__global__ __launch_bounds__(256) void cvt_kernel(const float* __restrict__ x,
                                                  _Float16* __restrict__ xT) {
    const int b  = blockIdx.z;
    const int hh = blockIdx.y;
    const int w0 = blockIdx.x * 32;
    __shared__ _Float16 t[32][CPAD];      // [w][cin], 36-dword lane stride

    const int wlane = threadIdx.x & 31;
    const int cg    = threadIdx.x >> 5;   // 0..7
    #pragma unroll
    for (int p = 0; p < 8; ++p) {
        int cin = p * 8 + cg;
        float v = x[(((size_t)b * CIN_ + cin) * H_ + hh) * W_ + w0 + wlane];
        t[wlane][cin] = (_Float16)v;
    }
    __syncthreads();

    const int w  = threadIdx.x >> 3;
    const int k8 = threadIdx.x & 7;
    v8h val = *(const v8h*)&t[w][k8 * 8];
    *(v8h*)(xT + (((size_t)b * H_ + hh) * W_ + w0 + w) * (size_t)CIN_ + k8 * 8) = val;
}

// ---------------------------------------------------------------------------
// Kernel 3 (fast path): implicit-GEMM conv reading f16 HWC xT.
// LDS window fill = 816 lane-level 128-bit block copies (async-to-LDS when
// the toolchain exposes GLOBAL_LOAD_ASYNC_TO_LDS_B128).
// ---------------------------------------------------------------------------
static __device__ __forceinline__ v16h cat8(v8h a, v8h b) {
    return __builtin_shufflevector(a, b, 0, 1, 2, 3, 4, 5, 6, 7,
                                         8, 9, 10, 11, 12, 13, 14, 15);
}

__global__ __launch_bounds__(128) void conv_kernel_hwc(
        const _Float16* __restrict__ xT,
        const _Float16* __restrict__ packedA,
        float* __restrict__ out) {
    const int b  = blockIdx.z;
    const int h  = blockIdx.y;
    const int w0 = blockIdx.x * 32;

    __shared__ _Float16 xs[3][34][CPAD];  // [row][col][cin], cin-innermost

    const int NT = 3 * 34 * 8;            // 816 x b128 transfers
    for (int idx = threadIdx.x; idx < NT; idx += 128) {
        int r    = idx / (34 * 8);
        int rem  = idx - r * (34 * 8);
        int col  = rem >> 3;
        int part = rem & 7;
        int hy = h - 1 + r;
        int wx = w0 - 1 + col;
        _Float16* dst = &xs[r][col][part * 8];
        if ((unsigned)hy < (unsigned)H_ && (unsigned)wx < (unsigned)W_) {
            const _Float16* src =
                xT + (((size_t)b * H_ + hy) * W_ + wx) * (size_t)CIN_ + part * 8;
#ifdef ASYNC_LDS
            __builtin_amdgcn_global_load_async_to_lds_b128(
                (gas1_t)(uintptr_t)src,
                (las3_t)(uint32_t)(uintptr_t)dst, 0, 0);
#else
            *(v8h*)dst = *(const v8h*)src;
#endif
        } else {
            *(v8h*)dst = (v8h){};          // zero padding at image borders
        }
    }
#ifdef ASYNC_LDS
#if __has_builtin(__builtin_amdgcn_s_wait_asynccnt)
    __builtin_amdgcn_s_wait_asynccnt(0);
#else
    asm volatile("s_wait_asynccnt 0x0" ::: "memory");
#endif
#endif
    __syncthreads();

    const int wave = threadIdx.x >> 5;
    const int lane = threadIdx.x & 31;
    const int n  = lane & 15;
    const int hi = lane >> 4;

    v8f acc0 = {};
    v8f acc1 = {};
    const _Float16* atiles = packedA + (((size_t)b * 4 + wave) * 9) * 2 * 512;

    #pragma unroll
    for (int dy = 0; dy < 3; ++dy) {
        #pragma unroll
        for (int dx = 0; dx < 3; ++dx) {
            #pragma unroll
            for (int ch = 0; ch < 2; ++ch) {
                const int tap = dy * 3 + dx;
                v16h a = *(const v16h*)(atiles + ((size_t)tap * 2 + ch) * 512 + lane * 16);
                const _Float16* p0 = &xs[dy][dx + n][ch * 32 + hi * 8];
                const _Float16* p1 = &xs[dy][dx + 16 + n][ch * 32 + hi * 8];
                v16h b0 = cat8(*(const v8h*)p0, *(const v8h*)(p0 + 16));
                v16h b1 = cat8(*(const v8h*)p1, *(const v8h*)(p1 + 16));
                acc0 = __builtin_amdgcn_wmma_f32_16x16x32_f16(
                        false, a, false, b0, (short)0, acc0, false, false);
                acc1 = __builtin_amdgcn_wmma_f32_16x16x32_f16(
                        false, a, false, b1, (short)0, acc1, false, false);
            }
        }
    }

    #pragma unroll
    for (int i = 0; i < 8; ++i) {
        int cout = wave * 16 + i + 8 * hi;
        size_t o = (((size_t)b * COUT_ + cout) * H_ + h) * W_ + w0;
        out[o + n]      = acc0[i];
        out[o + 16 + n] = acc1[i];
    }
}

// ---------------------------------------------------------------------------
// Kernel 3 (fallback): round-2 version reading f32 x directly.
// Used only when the workspace cannot hold xT.
// ---------------------------------------------------------------------------
__global__ __launch_bounds__(128) void conv_kernel_f32(
        const float* __restrict__ x,
        const _Float16* __restrict__ packedA,
        float* __restrict__ out) {
    const int b  = blockIdx.z;
    const int h  = blockIdx.y;
    const int w0 = blockIdx.x * 32;

    __shared__ _Float16 xs[3][34][CPAD];

    const int NE = 3 * CIN_ * 34;
    for (int e = threadIdx.x; e < NE; e += 128) {
        int r   = e / (CIN_ * 34);
        int rem = e - r * (CIN_ * 34);
        int cin = rem / 34;
        int col = rem - cin * 34;
        int hy = h - 1 + r;
        int wx = w0 - 1 + col;
        float v = 0.f;
        if ((unsigned)hy < (unsigned)H_ && (unsigned)wx < (unsigned)W_)
            v = x[(((size_t)b * CIN_ + cin) * H_ + hy) * W_ + wx];
        xs[r][col][cin] = (_Float16)v;
    }
    __syncthreads();

    const int wave = threadIdx.x >> 5;
    const int lane = threadIdx.x & 31;
    const int n  = lane & 15;
    const int hi = lane >> 4;

    v8f acc0 = {};
    v8f acc1 = {};
    const _Float16* atiles = packedA + (((size_t)b * 4 + wave) * 9) * 2 * 512;

    #pragma unroll
    for (int dy = 0; dy < 3; ++dy) {
        #pragma unroll
        for (int dx = 0; dx < 3; ++dx) {
            #pragma unroll
            for (int ch = 0; ch < 2; ++ch) {
                const int tap = dy * 3 + dx;
                v16h a = *(const v16h*)(atiles + ((size_t)tap * 2 + ch) * 512 + lane * 16);
                const _Float16* p0 = &xs[dy][dx + n][ch * 32 + hi * 8];
                const _Float16* p1 = &xs[dy][dx + 16 + n][ch * 32 + hi * 8];
                v16h b0 = cat8(*(const v8h*)p0, *(const v8h*)(p0 + 16));
                v16h b1 = cat8(*(const v8h*)p1, *(const v8h*)(p1 + 16));
                acc0 = __builtin_amdgcn_wmma_f32_16x16x32_f16(
                        false, a, false, b0, (short)0, acc0, false, false);
                acc1 = __builtin_amdgcn_wmma_f32_16x16x32_f16(
                        false, a, false, b1, (short)0, acc1, false, false);
            }
        }
    }

    #pragma unroll
    for (int i = 0; i < 8; ++i) {
        int cout = wave * 16 + i + 8 * hi;
        size_t o = (((size_t)b * COUT_ + cout) * H_ + h) * W_ + w0;
        out[o + n]      = acc0[i];
        out[o + 16 + n] = acc1[i];
    }
}

// ---------------------------------------------------------------------------
extern "C" void kernel_launch(void* const* d_in, const int* in_sizes, int n_in,
                              void* d_out, int out_size, void* d_ws, size_t ws_size,
                              hipStream_t stream) {
    const float* x      = (const float*)d_in[0];
    const float* w      = (const float*)d_in[1];
    const float* weight = (const float*)d_in[2];
    const float* mod_w  = (const float*)d_in[3];
    const float* mod_b  = (const float*)d_in[4];
    float* out = (float*)d_out;

    float*    style   = (float*)d_ws;
    _Float16* packedA = (_Float16*)((char*)d_ws + 4096);

    style_kernel<<<B_, CIN_, 0, stream>>>(w, mod_w, mod_b, style);
    pack_kernel<<<B_ * COUT_, CIN_, 0, stream>>>(weight, style, packedA);

    dim3 grid(W_ / 32, H_, B_);
    if (ws_size >= (size_t)XT_OFF + XT_BYTES) {
        _Float16* xT = (_Float16*)((char*)d_ws + XT_OFF);
        dim3 cgrid(W_ / 32, H_, B_);
        cvt_kernel<<<cgrid, 256, 0, stream>>>(x, xT);
        conv_kernel_hwc<<<grid, 128, 0, stream>>>(xT, packedA, out);
    } else {
        conv_kernel_f32<<<grid, 128, 0, stream>>>(x, packedA, out);
    }
}
